// MultiAttention_29592324670244
// MI455X (gfx1250) — matrix operand order repaired and checked by reading, not compiled, not run on hardware
//
#include <hip/hip_runtime.h>

// ---------------------------------------------------------------------------
// MI455X (gfx1250) multi-head causal attention, f16 WMMA with f32 accumulate.
// B=4, T=2048, D=1024, H=16, HD=VD=64.
// Data movement: GLOBAL_LOAD_ASYNC_TO_LDS_B128 (ASYNCcnt) when the toolchain
// exposes the builtin, register-double-buffered global->LDS otherwise.
// ---------------------------------------------------------------------------

typedef __attribute__((ext_vector_type(16))) _Float16 v16h;
typedef __attribute__((ext_vector_type(8)))  _Float16 v8h;
typedef __attribute__((ext_vector_type(4)))  _Float16 v4h;
typedef __attribute__((ext_vector_type(8)))  float    v8f;

union AF { v16h v; v8h h[2]; };

static constexpr int Bc  = 4;
static constexpr int Tc  = 2048;
static constexpr int Dc  = 1024;
static constexpr int Hc  = 16;
static constexpr int HDc = 64;
static constexpr int Mc  = Bc * Tc;          // 8192 rows for the big GEMMs
static constexpr int Nqkv = 3 * Hc * HDc;    // 3072

#if defined(__HIP_DEVICE_COMPILE__) && \
    __has_builtin(__builtin_amdgcn_global_load_async_to_lds_b128)
#define USE_ASYNC_LDS 1
#else
#define USE_ASYNC_LDS 0
#endif

#if USE_ASYNC_LDS
#define GASP __attribute__((address_space(1)))
#define LASP __attribute__((address_space(3)))
// The builtin wants vector_size(16) int pointers: AS(1) source, AS(3) dest.
typedef int vs4i __attribute__((vector_size(16)));
// Generic->AS casts via integers: generic LDS addr low 32 bits are the LDS
// offset; generic global pointers are numerically the AS(1) address.
__device__ __forceinline__ GASP vs4i* as_global(const void* p) {
  return (GASP vs4i*)(unsigned long long)p;
}
__device__ __forceinline__ LASP vs4i* as_lds(void* p) {
  return (LASP vs4i*)(unsigned int)(unsigned long long)p;
}
__device__ __forceinline__ void async_copy16(const void* g, void* l) {
  __builtin_amdgcn_global_load_async_to_lds_b128(as_global(g), as_lds(l), 0, 0);
}
__device__ __forceinline__ void wait_async0() {
#if __has_builtin(__builtin_amdgcn_s_wait_asynccnt)
  __builtin_amdgcn_s_wait_asynccnt(0);
#else
  asm volatile("s_wait_asynccnt 0" ::: "memory");
#endif
}
#endif

__device__ __forceinline__ v8f wmma16(v16h a, v16h b, v8f c) {
  // D = A(16x32 f16) * B(32x16 f16) + C(16x16 f32)
  return __builtin_amdgcn_wmma_f32_16x16x32_f16(
      /*neg_a=*/false, a, /*neg_b=*/false, b,
      /*c_mod=*/(short)0, c, /*reuse_a=*/false, /*reuse_b=*/false);
}

// Build a v16h fragment from two aligned 8-half runs.
__device__ __forceinline__ v16h frag_2x8(const _Float16* p0, const _Float16* p1) {
  AF u;
  u.h[0] = *(const v8h*)p0;
  u.h[1] = *(const v8h*)p1;
  return u.v;
}

// ---------------------------------------------------------------------------
// Pack kernels: f32 -> f16 conversions and weight transposes.
// ---------------------------------------------------------------------------

__global__ void pack_x_kernel(const float* __restrict__ x, _Float16* __restrict__ xb) {
  size_t i = ((size_t)blockIdx.x * 256 + threadIdx.x) * 4;
  float4 f = *(const float4*)(x + i);
  v4h h;
  h[0] = (_Float16)f.x; h[1] = (_Float16)f.y;
  h[2] = (_Float16)f.z; h[3] = (_Float16)f.w;
  *(v4h*)(xb + i) = h;
}

// wqkvt[n][d] with n = proj*1024 + h*64 + c  (N-major, K=D contiguous) and
// concatenated qkv bias (f32).
__global__ void pack_wqkv_kernel(const float* __restrict__ Wq, const float* __restrict__ Wk,
                                 const float* __restrict__ Wv,
                                 const float* __restrict__ bq, const float* __restrict__ bk,
                                 const float* __restrict__ bv,
                                 _Float16* __restrict__ wt, float* __restrict__ bcat) {
  int idx = blockIdx.x * 256 + threadIdx.x;   // < 3*1024*1024
  int n = idx >> 10;
  int d = idx & 1023;
  int proj = n >> 10;
  int h = (n >> 6) & 15;
  int c = n & 63;
  const float* W = (proj == 0) ? Wq : (proj == 1) ? Wk : Wv;  // (H, D, 64)
  wt[(size_t)n * Dc + d] = (_Float16)W[((size_t)h * Dc + d) * 64 + c];
  if (idx < Nqkv) {
    int p2 = idx >> 10;
    const float* bb = (p2 == 0) ? bq : (p2 == 1) ? bk : bv;
    bcat[idx] = bb[idx & 1023];
  }
}

// wot[n][k] = Wo[k][n]  (N-major for B-fragment loads)
__global__ void pack_wo_kernel(const float* __restrict__ Wo, _Float16* __restrict__ wot) {
  int idx = blockIdx.x * 256 + threadIdx.x;   // < 1024*1024
  int n = idx >> 10;
  int k = idx & 1023;
  wot[idx] = (_Float16)Wo[(size_t)k * Dc + n];
}

// ---------------------------------------------------------------------------
// Generic WMMA GEMM: C(M x N) = A(M x K, row-major f16) * Bt(N x K f16)^T + bias
// Block tile 128x128, 4 waves of 64x64 each, BK = 32, double-buffered LDS with
// one barrier per K-step.
// MODE 0: scatter f16 results into Q/K/V buffers laid out [b,h,t,c].
// MODE 1: f32 results row-major into fout.
// ---------------------------------------------------------------------------

template <int MODE>
__global__ void __launch_bounds__(128)
gemm_wmma_kernel(const _Float16* __restrict__ A, const _Float16* __restrict__ Bt,
                 const float* __restrict__ bias,
                 _Float16* __restrict__ qout, _Float16* __restrict__ kout,
                 _Float16* __restrict__ vout, float* __restrict__ fout) {
  constexpr int K  = Dc;   // 1024
  constexpr int LP = 40;   // LDS row pitch in halves (mult of 8 -> 16B aligned rows)
  __shared__ __align__(16) _Float16 aT[2][128 * LP];
  __shared__ __align__(16) _Float16 bT[2][128 * LP];

  const int tid  = threadIdx.x;
  const int lane = tid & 31;
  const int w    = tid >> 5;
  const int lm   = lane & 15;
  const int lh   = lane >> 4;
  const int m0   = blockIdx.x * 128;
  const int n0   = blockIdx.y * 128;
  const int wm   = (w >> 1) * 64;
  const int wn   = (w & 1) * 64;

  v8f acc[4][4];
#pragma unroll
  for (int i = 0; i < 4; i++)
#pragma unroll
    for (int j = 0; j < 4; j++)
#pragma unroll
      for (int p = 0; p < 8; p++) acc[i][j][p] = 0.0f;

  const _Float16* arow = A  + (size_t)(m0 + tid) * K;
  const _Float16* brow = Bt + (size_t)(n0 + tid) * K;

  auto compute_tile = [&](int buf) {
    v16h af[4], bf[4];
#pragma unroll
    for (int i = 0; i < 4; i++) {
      const _Float16* p = &aT[buf][(wm + i * 16 + lm) * LP + 8 * lh];
      af[i] = frag_2x8(p, p + 16);
    }
#pragma unroll
    for (int j = 0; j < 4; j++) {
      const _Float16* p = &bT[buf][(wn + j * 16 + lm) * LP + 8 * lh];
      bf[j] = frag_2x8(p, p + 16);
    }
#pragma unroll
    for (int i = 0; i < 4; i++)
#pragma unroll
      for (int j = 0; j < 4; j++)
        acc[i][j] = wmma16(af[i], bf[j], acc[i][j]);
  };

#if USE_ASYNC_LDS
  // --- ASYNCcnt-tracked global->LDS DMA path, one barrier per K-step. ---
  auto issue_tile = [&](int k0, int buf) {
    const _Float16* ga = arow + k0;
    const _Float16* gb = brow + k0;
    _Float16* la = &aT[buf][tid * LP];
    _Float16* lb = &bT[buf][tid * LP];
#pragma unroll
    for (int c = 0; c < 4; c++) {
      async_copy16(ga + 8 * c, la + 8 * c);
      async_copy16(gb + 8 * c, lb + 8 * c);
    }
  };
  issue_tile(0, 0);
  int cur = 0;
  for (int k0 = 0; k0 < K; k0 += 32) {
    wait_async0();        // my async writes for buffer `cur` are in LDS
    __syncthreads();      // everyone's are
    if (k0 + 32 < K) issue_tile(k0 + 32, cur ^ 1);
    compute_tile(cur);
    cur ^= 1;
  }
#else
  // --- Fallback: register-double-buffered staging (loads overlap WMMA). ---
  v8h ra[4], rb[4];
  {
    const v8h* sa = (const v8h*)arow;
    const v8h* sb = (const v8h*)brow;
#pragma unroll
    for (int c = 0; c < 4; c++) { ra[c] = sa[c]; rb[c] = sb[c]; }
    v8h* da = (v8h*)&aT[0][tid * LP];
    v8h* db = (v8h*)&bT[0][tid * LP];
#pragma unroll
    for (int c = 0; c < 4; c++) { da[c] = ra[c]; db[c] = rb[c]; }
  }
  int cur = 0;
  for (int k0 = 0; k0 < K; k0 += 32) {
    __syncthreads();
    if (k0 + 32 < K) {
      const v8h* sa = (const v8h*)(arow + k0 + 32);
      const v8h* sb = (const v8h*)(brow + k0 + 32);
#pragma unroll
      for (int c = 0; c < 4; c++) { ra[c] = sa[c]; rb[c] = sb[c]; }
    }
    compute_tile(cur);
    if (k0 + 32 < K) {
      v8h* da = (v8h*)&aT[cur ^ 1][tid * LP];
      v8h* db = (v8h*)&bT[cur ^ 1][tid * LP];
#pragma unroll
      for (int c = 0; c < 4; c++) { da[c] = ra[c]; db[c] = rb[c]; }
    }
    cur ^= 1;
  }
#endif

  // Epilogue: C-layout element p of lane -> row p + 8*lh, col lm.
#pragma unroll
  for (int i = 0; i < 4; i++)
#pragma unroll
    for (int j = 0; j < 4; j++)
#pragma unroll
      for (int p = 0; p < 8; p++) {
        int m = m0 + wm + i * 16 + p + 8 * lh;
        int n = n0 + wn + j * 16 + lm;
        float val = acc[i][j][p] + bias[n];
        if (MODE == 0) {
          int proj = n >> 10;
          int hh   = (n >> 6) & 15;
          int c    = n & 63;
          int bidx = m >> 11;      // m = b*T + t, T = 2048
          int t    = m & 2047;
          size_t dst = (((size_t)(bidx * Hc + hh)) * Tc + t) * 64 + c;
          _Float16* o = (proj == 0) ? qout : (proj == 1) ? kout : vout;
          o[dst] = (_Float16)val;
        } else {
          fout[(size_t)m * Dc + n] = val;
        }
      }
}

// ---------------------------------------------------------------------------
// Flash-style causal attention. Grid: (B*H, T/256). 4 waves; each wave owns a
// 64-query block, workgroup shares K/V tiles of 64 keys.
// ---------------------------------------------------------------------------

__global__ void __launch_bounds__(128)
attn_wmma_kernel(const _Float16* __restrict__ Q, const _Float16* __restrict__ Kmat,
                 const _Float16* __restrict__ Vmat, _Float16* __restrict__ concat) {
  constexpr int LP = 72;  // LDS pitch in halves
  __shared__ __align__(16) _Float16 kt[64 * LP];        // [key][c]
  __shared__ __align__(16) _Float16 vt[64 * LP];        // [vchan][key]
  __shared__ __align__(16) _Float16 pt[4][64 * LP];     // per-wave P [m][key]

  const int tid  = threadIdx.x;
  const int lane = tid & 31;
  const int w    = tid >> 5;
  const int lm   = lane & 15;
  const int lh   = lane >> 4;
  const int bh   = blockIdx.x;
  const int bidx = bh >> 4;
  const int hidx = bh & 15;
  const int q0   = blockIdx.y * 256 + w * 64;

  const _Float16* qbase = Q    + (size_t)bh * Tc * 64;
  const _Float16* kbase = Kmat + (size_t)bh * Tc * 64;
  const _Float16* vbase = Vmat + (size_t)bh * Tc * 64;
  _Float16* pw = &pt[w][0];

  // Q fragments held in registers across the whole key loop.
  v16h qf[4][2];
#pragma unroll
  for (int i = 0; i < 4; i++)
#pragma unroll
    for (int kc = 0; kc < 2; kc++) {
      const _Float16* p = qbase + (size_t)(q0 + i * 16 + lm) * 64 + kc * 32 + 8 * lh;
      qf[i][kc] = frag_2x8(p, p + 16);
    }

  v8f   o[4][4];
  float mrow[4][8], lrow[4][8];
#pragma unroll
  for (int i = 0; i < 4; i++) {
#pragma unroll
    for (int p = 0; p < 8; p++) { mrow[i][p] = -1e30f; lrow[i][p] = 0.0f; }
#pragma unroll
    for (int j = 0; j < 4; j++)
#pragma unroll
      for (int p = 0; p < 8; p++) o[i][j][p] = 0.0f;
  }

  const int nkb = blockIdx.y * 4 + 4;  // key blocks needed to cover the diagonal
  for (int kb = 0; kb < nkb; kb++) {
    const int kb0 = kb * 64;

    // Stage K tile (row-major [key][c], the Bt layout for Q*K^T).
    {
      int r  = tid >> 1;
      int cc = (tid & 1) * 32;
      const _Float16* g = kbase + (size_t)(kb0 + r) * 64 + cc;
      _Float16* l = kt + r * LP + cc;
#if USE_ASYNC_LDS
#pragma unroll
      for (int c = 0; c < 4; c++) async_copy16(g + 8 * c, l + 8 * c);
#else
      const v8h* s = (const v8h*)g;
      v8h x0 = s[0], x1 = s[1], x2 = s[2], x3 = s[3];
      v8h* dd = (v8h*)l;
      dd[0] = x0; dd[1] = x1; dd[2] = x2; dd[3] = x3;
#endif
    }
    // Stage V transposed: vt[vchan][key] (Bt layout for P*V).
    for (int idx = tid; idx < 4096; idx += 128) {
      int key = idx >> 6, vv = idx & 63;
      vt[vv * LP + key] = vbase[(size_t)(kb0 + key) * 64 + vv];
    }
#if USE_ASYNC_LDS
    wait_async0();
#endif
    __syncthreads();

    if (kb0 <= q0 + 63) {  // wave-uniform: skip fully-masked key blocks
      // S = Q * K^T (64x64), f32 accumulators.
      v8f sc[4][4];
#pragma unroll
      for (int i = 0; i < 4; i++)
#pragma unroll
        for (int j = 0; j < 4; j++)
#pragma unroll
          for (int p = 0; p < 8; p++) sc[i][j][p] = 0.0f;

#pragma unroll
      for (int kc = 0; kc < 2; kc++) {
        v16h bf[4];
#pragma unroll
        for (int j = 0; j < 4; j++) {
          const _Float16* p = kt + (j * 16 + lm) * LP + kc * 32 + 8 * lh;
          bf[j] = frag_2x8(p, p + 16);
        }
#pragma unroll
        for (int i = 0; i < 4; i++)
#pragma unroll
          for (int j = 0; j < 4; j++)
            sc[i][j] = wmma16(qf[i][kc], bf[j], sc[i][j]);
      }

      // Scale, causal mask, online softmax update.
#pragma unroll
      for (int i = 0; i < 4; i++) {
#pragma unroll
        for (int p = 0; p < 8; p++) {
          int row = q0 + i * 16 + p + 8 * lh;
          float mx = -1e30f;
#pragma unroll
          for (int j = 0; j < 4; j++) {
            int col = kb0 + j * 16 + lm;
            float sv = sc[i][j][p] * 0.125f;        // HD^-0.5
            sv = (col > row) ? -1e30f : sv;
            sc[i][j][p] = sv;
            mx = fmaxf(mx, sv);
          }
#pragma unroll
          for (int off = 1; off < 16; off <<= 1)
            mx = fmaxf(mx, __shfl_xor(mx, off, 32));
          float mold = mrow[i][p];
          float mnew = fmaxf(mold, mx);
          float alpha = __expf(mold - mnew);
          mrow[i][p] = mnew;
          float rs = 0.0f;
#pragma unroll
          for (int j = 0; j < 4; j++) {
            float e = __expf(sc[i][j][p] - mnew);
            sc[i][j][p] = e;
            rs += e;
          }
#pragma unroll
          for (int off = 1; off < 16; off <<= 1)
            rs += __shfl_xor(rs, off, 32);
          lrow[i][p] = lrow[i][p] * alpha + rs;
#pragma unroll
          for (int j = 0; j < 4; j++) o[i][j][p] *= alpha;
        }
      }

      // Round-trip P through per-wave LDS: C-layout f32 -> row-major f16.
#pragma unroll
      for (int i = 0; i < 4; i++)
#pragma unroll
        for (int j = 0; j < 4; j++)
#pragma unroll
          for (int p = 0; p < 8; p++)
            pw[(i * 16 + p + 8 * lh) * LP + j * 16 + lm] = (_Float16)sc[i][j][p];

      // O += P * V
#pragma unroll
      for (int kc = 0; kc < 2; kc++) {
        v16h pf[4], vf[4];
#pragma unroll
        for (int i = 0; i < 4; i++) {
          const _Float16* p = pw + (i * 16 + lm) * LP + kc * 32 + 8 * lh;
          pf[i] = frag_2x8(p, p + 16);
        }
#pragma unroll
        for (int j = 0; j < 4; j++) {
          const _Float16* p = vt + (j * 16 + lm) * LP + kc * 32 + 8 * lh;
          vf[j] = frag_2x8(p, p + 16);
        }
#pragma unroll
        for (int i = 0; i < 4; i++)
#pragma unroll
          for (int j = 0; j < 4; j++)
            o[i][j] = wmma16(pf[i], vf[j], o[i][j]);
      }
    }
    __syncthreads();
  }

  // Normalize and write concat[b][t][h*64 + v] as f16 for the output GEMM.
#pragma unroll
  for (int i = 0; i < 4; i++)
#pragma unroll
    for (int p = 0; p < 8; p++) {
      float inv = 1.0f / lrow[i][p];
      int t = q0 + i * 16 + p + 8 * lh;
#pragma unroll
      for (int j = 0; j < 4; j++) {
        int col = hidx * 64 + j * 16 + lm;
        concat[((size_t)bidx * Tc + t) * Dc + col] = (_Float16)(o[i][j][p] * inv);
      }
    }
}

// ---------------------------------------------------------------------------
// Host launcher.
// ---------------------------------------------------------------------------

extern "C" void kernel_launch(void* const* d_in, const int* in_sizes, int n_in,
                              void* d_out, int out_size, void* d_ws, size_t ws_size,
                              hipStream_t stream) {
  (void)in_sizes; (void)n_in; (void)out_size; (void)ws_size;

  const float* x  = (const float*)d_in[0];
  const float* Wq = (const float*)d_in[1];
  const float* bq = (const float*)d_in[2];
  const float* Wk = (const float*)d_in[3];
  const float* bk = (const float*)d_in[4];
  const float* Wv = (const float*)d_in[5];
  const float* bv = (const float*)d_in[6];
  const float* Wo = (const float*)d_in[7];
  const float* bo = (const float*)d_in[8];
  float* out = (float*)d_out;

  char* wsp = (char*)d_ws;
  size_t off = 0;
  auto take = [&](size_t bytes) -> char* {
    char* p = wsp + off;
    off += (bytes + 255) & ~(size_t)255;
    return p;
  };

  _Float16* xb    = (_Float16*)take((size_t)Mc * Dc * sizeof(_Float16));
  _Float16* wqkvt = (_Float16*)take((size_t)Nqkv * Dc * sizeof(_Float16));
  float*    bqkv  = (float*)take((size_t)Nqkv * sizeof(float));
  _Float16* wot   = (_Float16*)take((size_t)Dc * Dc * sizeof(_Float16));
  _Float16* qbuf  = (_Float16*)take((size_t)Bc * Hc * Tc * HDc * sizeof(_Float16));
  _Float16* kbuf  = (_Float16*)take((size_t)Bc * Hc * Tc * HDc * sizeof(_Float16));
  _Float16* vbuf  = (_Float16*)take((size_t)Bc * Hc * Tc * HDc * sizeof(_Float16));
  _Float16* cbuf  = (_Float16*)take((size_t)Mc * Dc * sizeof(_Float16));

  pack_x_kernel<<<(Mc * Dc) / 4 / 256, 256, 0, stream>>>(x, xb);
  pack_wqkv_kernel<<<(3 * 1024 * 1024) / 256, 256, 0, stream>>>(Wq, Wk, Wv, bq, bk, bv,
                                                                wqkvt, bqkv);
  pack_wo_kernel<<<(1024 * 1024) / 256, 256, 0, stream>>>(Wo, wot);

  gemm_wmma_kernel<0><<<dim3(Mc / 128, Nqkv / 128), 128, 0, stream>>>(
      xb, wqkvt, bqkv, qbuf, kbuf, vbuf, nullptr);

  attn_wmma_kernel<<<dim3(Bc * Hc, Tc / 256), 128, 0, stream>>>(qbuf, kbuf, vbuf, cbuf);

  gemm_wmma_kernel<1><<<dim3(Mc / 128, Dc / 128), 128, 0, stream>>>(
      cbuf, wot, bo, nullptr, nullptr, nullptr, out);
}
